// GATv2GraphNet_IMDB_34832184770979
// MI455X (gfx1250) — matrix operand order repaired and checked by reading, not compiled, not run on hardware
//
#include <hip/hip_runtime.h>
#include <hip/hip_bf16.h>
#include <math.h>

#define NN   50000
#define EE   1600000
#define ESL  (EE + NN)       // edges + self loops
#define HH   2
#define CC   16
#define HC   32
#define GG   512
#define NEG_SLOPE 0.2f
#define NTILES (NN / 16)     // 3125 exactly

typedef __attribute__((ext_vector_type(16))) _Float16 v16h;
typedef __attribute__((ext_vector_type(8)))  _Float16 v8h;
typedef __attribute__((ext_vector_type(8)))  float    v8f;

// ---------------- utility ----------------

__device__ __forceinline__ void atomicMaxF(float* addr, float v) {
    // monotonic int trick: works across sign boundary when init = -inf
    if (v >= 0.0f) atomicMax((int*)addr, __float_as_int(v));
    else           atomicMin((unsigned int*)addr, (unsigned int)__float_as_int(v));
}

__global__ void k_fill_f32(float* p, float val, int n) {
    int i = blockIdx.x * blockDim.x + threadIdx.x;
    if (i < n) p[i] = val;
}

// ---------------- degree (original edges only) ----------------

__global__ void k_degree(const int* __restrict__ ei, float* __restrict__ deg) {
    int e = blockIdx.x * blockDim.x + threadIdx.x;
    if (e >= EE) return;
    atomicAdd(&deg[ei[e]],      1.0f);
    atomicAdd(&deg[ei[EE + e]], 1.0f);
}

// ---------------- conv1 linear: x=[1,deg,rand] @ W^T + b  (K=3, scalar) --------

__global__ void k_conv1_lin(const float* __restrict__ deg, const float* __restrict__ rnd,
                            const float* __restrict__ Wl, const float* __restrict__ bl,
                            const float* __restrict__ Wr, const float* __restrict__ br,
                            float* __restrict__ xl, float* __restrict__ xr) {
    int i = blockIdx.x * blockDim.x + threadIdx.x;
    if (i >= NN) return;
    float x1 = deg[i], x2 = rnd[i];
    #pragma unroll
    for (int o = 0; o < HC; ++o) {
        xl[i * HC + o] = Wl[o * 3] + Wl[o * 3 + 1] * x1 + Wl[o * 3 + 2] * x2 + bl[o];
        xr[i * HC + o] = Wr[o * 3] + Wr[o * 3 + 1] * x1 + Wr[o * 3 + 2] * x2 + br[o];
    }
}

// ---------------- weight f32 -> f16 ----------------

__global__ void k_cvt_w(const float* __restrict__ w2l, const float* __restrict__ w2r,
                        _Float16* __restrict__ wl16, _Float16* __restrict__ wr16) {
    int i = blockIdx.x * blockDim.x + threadIdx.x;
    if (i < HC * HC) {
        wl16[i] = (_Float16)w2l[i];
        wr16[i] = (_Float16)w2r[i];
    }
}

// ---------------- conv2 linear via WMMA: out = x(16x32) @ W^T(32x32) + b -------
// A frag (16x32 f16): row = lane&15, K = t + 16*(lane>>4) -> contiguous 16 halfs of x[row][*]
// B frag (32x16 f16): col = lane&15, K = t + 16*(lane>>4)  -> contiguous 16 halfs of W[col][*]
// D (16x16 f32): element (M = r + 8*(lane>>4), N = lane&15) in VGPR r

__global__ __launch_bounds__(256) void k_lin32_wmma(
        const _Float16* __restrict__ x16,   // NN x 32, f16 row-major
        const _Float16* __restrict__ wl,    // 32x32 row-major (out,in)
        const float*    __restrict__ bl,
        const _Float16* __restrict__ wr,
        const float*    __restrict__ br,
        float* __restrict__ outl,           // NN x 32
        float* __restrict__ outr) {
    int wave = threadIdx.x >> 5;
    int lane = threadIdx.x & 31;
    int tile = blockIdx.x * 8 + wave;
    if (tile >= NTILES) return;           // uniform per wave -> EXEC all 1s inside
    int lm = lane & 15;
    int hi = lane >> 4;                   // 0 or 1

    // A fragment: two 16B loads
    const _Float16* rowp = x16 + (size_t)(tile * 16 + lm) * HC;
    int kb = hi * 16;
    union { v16h v; _Float16 e[16]; } A;
    *(v8h*)&A.e[0] = *(const v8h*)(rowp + kb);
    *(v8h*)&A.e[8] = *(const v8h*)(rowp + kb + 8);

    int kb2 = hi * 16;
    #pragma unroll
    for (int cb = 0; cb < 2; ++cb) {      // output column block 0..15 / 16..31
        int n = lm + cb * 16;
        v16h Bl = *(const v16h*)(wl + n * HC + kb2);
        v16h Br = *(const v16h*)(wr + n * HC + kb2);
        v8f Cl, Cr;
        float cl = bl[n], cr = br[n];
        #pragma unroll
        for (int r = 0; r < 8; ++r) { Cl[r] = cl; Cr[r] = cr; }
        v8f Dl = __builtin_amdgcn_wmma_f32_16x16x32_f16(false, A.v, false, Bl,
                                                        (short)0, Cl, false, false);
        v8f Dr = __builtin_amdgcn_wmma_f32_16x16x32_f16(false, A.v, false, Br,
                                                        (short)0, Cr, false, false);
        #pragma unroll
        for (int r = 0; r < 8; ++r) {
            int m = tile * 16 + r + hi * 8;
            outl[(size_t)m * HC + n] = Dl[r];
            outr[(size_t)m * HC + n] = Dr[r];
        }
    }
}

// ---------------- edge pass 1: score + segment max ----------------

__global__ void k_edge_score(const int* __restrict__ ei,
                             const float* __restrict__ xl, const float* __restrict__ xr,
                             const float* __restrict__ att,
                             float* __restrict__ scores, float* __restrict__ smax) {
    int e = blockIdx.x * blockDim.x + threadIdx.x;
    if (e >= ESL) return;
    int s, d;
    if (e < EE) { s = ei[e]; d = ei[EE + e]; } else { s = e - EE; d = s; }
    const float* xs = xl + (size_t)s * HC;
    const float* xd = xr + (size_t)d * HC;
    #pragma unroll
    for (int h = 0; h < HH; ++h) {
        float sc = 0.0f;
        #pragma unroll
        for (int q = 0; q < 4; ++q) {
            float4 a = *(const float4*)(xs + h * CC + q * 4);
            float4 b = *(const float4*)(xd + h * CC + q * 4);
            float4 t = *(const float4*)(att + h * CC + q * 4);
            float m;
            m = a.x + b.x; m = m > 0.f ? m : NEG_SLOPE * m; sc += m * t.x;
            m = a.y + b.y; m = m > 0.f ? m : NEG_SLOPE * m; sc += m * t.y;
            m = a.z + b.z; m = m > 0.f ? m : NEG_SLOPE * m; sc += m * t.z;
            m = a.w + b.w; m = m > 0.f ? m : NEG_SLOPE * m; sc += m * t.w;
        }
        scores[(size_t)e * HH + h] = sc;
        atomicMaxF(&smax[(size_t)d * HH + h], sc);
    }
}

// ---------------- edge pass 2: exp + denom + unnormalized weighted sum --------

__global__ void k_edge_accum(const int* __restrict__ ei,
                             const float* __restrict__ xl,
                             const float* __restrict__ scores, const float* __restrict__ smax,
                             float* __restrict__ denom, float* __restrict__ accum) {
    int e = blockIdx.x * blockDim.x + threadIdx.x;
    if (e >= ESL) return;
    int s, d;
    if (e < EE) { s = ei[e]; d = ei[EE + e]; } else { s = e - EE; d = s; }
    #pragma unroll
    for (int h = 0; h < HH; ++h) {
        float ex = expf(scores[(size_t)e * HH + h] - smax[(size_t)d * HH + h]);
        atomicAdd(&denom[(size_t)d * HH + h], ex);
        const float* xs = xl + (size_t)s * HC + h * CC;
        float* ap = accum + (size_t)d * HC + h * CC;
        #pragma unroll
        for (int c = 0; c < CC; ++c) atomicAdd(&ap[c], ex * xs[c]);
    }
}

// ---------------- finalize: normalize + bias + ELU ----------------
// mode 0: write f16 (feeds WMMA of conv2); mode 1: write f32 (feeds pooling)

__global__ void k_finalize(const float* __restrict__ accum, const float* __restrict__ denom,
                           const float* __restrict__ bias,
                           _Float16* __restrict__ outh, float* __restrict__ outf, int mode) {
    int i = blockIdx.x * blockDim.x + threadIdx.x;
    if (i >= NN) return;
    #pragma unroll
    for (int j = 0; j < HC; ++j) {
        int h = j >> 4;
        float v = accum[(size_t)i * HC + j] / (denom[(size_t)i * HH + h] + 1e-16f) + bias[j];
        v = v > 0.f ? v : (expf(v) - 1.0f);   // ELU
        if (mode == 0) outh[(size_t)i * HC + j] = (_Float16)v;
        else           outf[(size_t)i * HC + j] = v;
    }
}

// ---------------- global mean pool ----------------

__global__ void k_pool(const int* __restrict__ batch, const float* __restrict__ x,
                       float* __restrict__ pooled, float* __restrict__ counts) {
    int i = blockIdx.x * blockDim.x + threadIdx.x;
    if (i >= NN) return;
    int g = batch[i];
    atomicAdd(&counts[g], 1.0f);
    const float* xp = x + (size_t)i * HC;
    float* pp = pooled + (size_t)g * HC;
    #pragma unroll
    for (int j = 0; j < HC; ++j) atomicAdd(&pp[j], xp[j]);
}

// ---------------- fc + log_softmax ----------------

__global__ void k_head(const float* __restrict__ pooled, const float* __restrict__ counts,
                       const float* __restrict__ Wfc, const float* __restrict__ bfc,
                       float* __restrict__ out) {
    int g = blockIdx.x * blockDim.x + threadIdx.x;
    if (g >= GG) return;
    float cnt = counts[g]; cnt = cnt < 1.0f ? 1.0f : cnt;
    float l0 = bfc[0], l1 = bfc[1];
    #pragma unroll
    for (int j = 0; j < HC; ++j) {
        float p = pooled[(size_t)g * HC + j] / cnt;
        l0 += p * Wfc[j];
        l1 += p * Wfc[HC + j];
    }
    float m = fmaxf(l0, l1);
    float lse = m + logf(expf(l0 - m) + expf(l1 - m));
    out[g * 2 + 0] = l0 - lse;
    out[g * 2 + 1] = l1 - lse;
}

// ---------------- launch ----------------

static inline size_t al256(size_t x) { return (x + 255) & ~(size_t)255; }

extern "C" void kernel_launch(void* const* d_in, const int* in_sizes, int n_in,
                              void* d_out, int out_size, void* d_ws, size_t ws_size,
                              hipStream_t stream) {
    const int*   ei    = (const int*)  d_in[0];
    const int*   batch = (const int*)  d_in[1];
    const float* rnd   = (const float*)d_in[2];
    const float* W1l   = (const float*)d_in[3];
    const float* b1l   = (const float*)d_in[4];
    const float* W1r   = (const float*)d_in[5];
    const float* b1r   = (const float*)d_in[6];
    const float* att1  = (const float*)d_in[7];
    const float* bias1 = (const float*)d_in[8];
    const float* W2l   = (const float*)d_in[9];
    const float* b2l   = (const float*)d_in[10];
    const float* W2r   = (const float*)d_in[11];
    const float* b2r   = (const float*)d_in[12];
    const float* att2  = (const float*)d_in[13];
    const float* bias2 = (const float*)d_in[14];
    const float* Wfc   = (const float*)d_in[15];
    const float* bfc   = (const float*)d_in[16];
    float* out = (float*)d_out;

    // workspace layout
    char* w = (char*)d_ws;
    size_t o = 0;
    float*    deg    = (float*)(w + o);  o += al256((size_t)NN * 4);
    float*    xl     = (float*)(w + o);  o += al256((size_t)NN * HC * 4);
    float*    xr     = (float*)(w + o);  o += al256((size_t)NN * HC * 4);
    float*    accum  = (float*)(w + o);  o += al256((size_t)NN * HC * 4);
    float*    denom  = (float*)(w + o);  o += al256((size_t)NN * HH * 4);
    float*    smax   = (float*)(w + o);  o += al256((size_t)NN * HH * 4);
    float*    scores = (float*)(w + o);  o += al256((size_t)ESL * HH * 4);
    _Float16* h1h    = (_Float16*)(w + o); o += al256((size_t)NN * HC * 2);
    float*    h2     = (float*)(w + o);  o += al256((size_t)NN * HC * 4);
    _Float16* wl16   = (_Float16*)(w + o); o += al256((size_t)HC * HC * 2);
    _Float16* wr16   = (_Float16*)(w + o); o += al256((size_t)HC * HC * 2);
    float*    pooled = (float*)(w + o);  o += al256((size_t)GG * HC * 4);
    float*    counts = (float*)(w + o);  o += al256((size_t)GG * 4);
    (void)ws_size; (void)n_in; (void)in_sizes; (void)out_size;

    const int TB = 256;
    dim3 blk(TB);
    dim3 gN((NN + TB - 1) / TB);
    dim3 gE((EE + TB - 1) / TB);
    dim3 gES((ESL + TB - 1) / TB);
    dim3 gWMMA((NTILES + 7) / 8);
    const float NEGINF = -__builtin_huge_valf();

    // degree features
    k_fill_f32<<<gN, blk, 0, stream>>>(deg, 0.0f, NN);
    k_degree<<<gE, blk, 0, stream>>>(ei, deg);

    // ---- conv1 ----
    k_conv1_lin<<<gN, blk, 0, stream>>>(deg, rnd, W1l, b1l, W1r, b1r, xl, xr);
    k_fill_f32<<<dim3((NN * HH + TB - 1) / TB), blk, 0, stream>>>(smax, NEGINF, NN * HH);
    k_fill_f32<<<dim3((NN * HH + TB - 1) / TB), blk, 0, stream>>>(denom, 0.0f, NN * HH);
    k_fill_f32<<<dim3((NN * HC + TB - 1) / TB), blk, 0, stream>>>(accum, 0.0f, NN * HC);
    k_edge_score<<<gES, blk, 0, stream>>>(ei, xl, xr, att1, scores, smax);
    k_edge_accum<<<gES, blk, 0, stream>>>(ei, xl, scores, smax, denom, accum);
    k_finalize<<<gN, blk, 0, stream>>>(accum, denom, bias1, h1h, (float*)nullptr, 0);

    // ---- conv2 ----
    k_cvt_w<<<dim3((HC * HC + TB - 1) / TB), blk, 0, stream>>>(W2l, W2r, wl16, wr16);
    k_lin32_wmma<<<gWMMA, blk, 0, stream>>>(h1h, wl16, b2l, wr16, b2r, xl, xr);
    k_fill_f32<<<dim3((NN * HH + TB - 1) / TB), blk, 0, stream>>>(smax, NEGINF, NN * HH);
    k_fill_f32<<<dim3((NN * HH + TB - 1) / TB), blk, 0, stream>>>(denom, 0.0f, NN * HH);
    k_fill_f32<<<dim3((NN * HC + TB - 1) / TB), blk, 0, stream>>>(accum, 0.0f, NN * HC);
    k_edge_score<<<gES, blk, 0, stream>>>(ei, xl, xr, att2, scores, smax);
    k_edge_accum<<<gES, blk, 0, stream>>>(ei, xl, scores, smax, denom, accum);
    k_finalize<<<gN, blk, 0, stream>>>(accum, denom, bias2, (_Float16*)nullptr, h2, 1);

    // ---- pool + head ----
    k_fill_f32<<<dim3((GG * HC + TB - 1) / TB), blk, 0, stream>>>(pooled, 0.0f, GG * HC);
    k_fill_f32<<<dim3((GG + TB - 1) / TB), blk, 0, stream>>>(counts, 0.0f, GG);
    k_pool<<<gN, blk, 0, stream>>>(batch, h2, pooled, counts);
    k_head<<<dim3((GG + TB - 1) / TB), blk, 0, stream>>>(pooled, counts, Wfc, bfc, out);
}
